// MoELayer_55121610277149
// MI455X (gfx1250) — matrix operand order repaired and checked by reading, not compiled, not run on hardware
//
#include <hip/hip_runtime.h>
#include <hip/hip_bf16.h>

// ---------------- problem constants ----------------
#define T_TOK 8192      // 4 * 2048 tokens
#define DMODEL 1024
#define DFF 1024
#define NEXP 8
#define ACT_STRIDE 1032 // DFF + 8 halves pad -> conflict-free ds_load_b128

typedef __attribute__((ext_vector_type(16))) __bf16 v16bf;
typedef __attribute__((ext_vector_type(8)))  float  v8f;
typedef __attribute__((ext_vector_type(4)))  unsigned int u32x4;

union Frag { v16bf v; u32x4 u[2]; };

// ---------------- utility kernels ----------------
__global__ void zero_kernel(float* __restrict__ p, int n) {
    int i = blockIdx.x * blockDim.x + threadIdx.x;
    int stride = gridDim.x * blockDim.x;
    for (; i < n; i += stride) p[i] = 0.0f;
}

__global__ void cvt_bf16_kernel(__bf16* __restrict__ dst, const float* __restrict__ src, int n) {
    int i = blockIdx.x * blockDim.x + threadIdx.x;
    int stride = gridDim.x * blockDim.x;
    for (; i < n; i += stride) dst[i] = (__bf16)src[i];
}

// ---------------- router: one wave32 per token ----------------
__global__ __launch_bounds__(256) void router_kernel(
    const float* __restrict__ x, const float* __restrict__ gw,
    float* __restrict__ probs, int* __restrict__ comb_e, float* __restrict__ comb_w)
{
    const int wv   = threadIdx.x >> 5;
    const int lane = threadIdx.x & 31;
    const int t    = blockIdx.x * 8 + wv;          // grid = T/8 blocks
    const float* xr = x + (size_t)t * DMODEL;

    float p[NEXP];
#pragma unroll
    for (int e = 0; e < NEXP; ++e) p[e] = 0.0f;

    for (int d = lane; d < DMODEL; d += 32) {
        float xv = xr[d];
#pragma unroll
        for (int e = 0; e < NEXP; ++e) p[e] += xv * gw[e * DMODEL + d];
    }
#pragma unroll
    for (int e = 0; e < NEXP; ++e)
        for (int off = 16; off > 0; off >>= 1) p[e] += __shfl_xor(p[e], off);

    if (lane == 0) {
        float mx = p[0];
#pragma unroll
        for (int e = 1; e < NEXP; ++e) mx = fmaxf(mx, p[e]);
        float pr[NEXP]; float s = 0.0f;
#pragma unroll
        for (int e = 0; e < NEXP; ++e) { pr[e] = __expf(p[e] - mx); s += pr[e]; }
        float inv = 1.0f / s;
#pragma unroll
        for (int e = 0; e < NEXP; ++e) { pr[e] *= inv; probs[(size_t)t * NEXP + e] = pr[e]; }
        int i0 = 0; float v0 = pr[0];
#pragma unroll
        for (int e = 1; e < NEXP; ++e) if (pr[e] > v0) { v0 = pr[e]; i0 = e; }
        int i1 = -1; float v1 = -1.0f;
#pragma unroll
        for (int e = 0; e < NEXP; ++e) if (e != i0 && pr[e] > v1) { v1 = pr[e]; i1 = e; }
        float sw = 1.0f / (v0 + v1);
        comb_e[t * 2 + 0] = i0; comb_e[t * 2 + 1] = i1;
        comb_w[t * 2 + 0] = v0 * sw; comb_w[t * 2 + 1] = v1 * sw;
    }
}

// ---------------- deterministic per-expert compaction ----------------
__global__ __launch_bounds__(256) void compact_kernel(
    const int* __restrict__ comb_e, const float* __restrict__ comb_w,
    int* __restrict__ tok_idx, float* __restrict__ tok_w, int* __restrict__ cnt)
{
    const int e = blockIdx.x;
    const int tid = threadIdx.x;
    __shared__ int scan[256];
    __shared__ int base_s;
    if (tid == 0) base_s = 0;
    __syncthreads();

    for (int c = 0; c < T_TOK; c += 256) {
        int t = c + tid;
        int flag = 0; float w = 0.0f;
        int e0 = comb_e[t * 2], e1 = comb_e[t * 2 + 1];
        if (e0 == e)      { flag = 1; w = comb_w[t * 2]; }
        else if (e1 == e) { flag = 1; w = comb_w[t * 2 + 1]; }
        scan[tid] = flag;
        __syncthreads();
        for (int off = 1; off < 256; off <<= 1) {
            int v = (tid >= off) ? scan[tid - off] : 0;
            __syncthreads();
            scan[tid] += v;
            __syncthreads();
        }
        int pos = scan[tid] - flag;
        int total = scan[255];
        int base = base_s;
        if (flag) {
            tok_idx[(size_t)e * T_TOK + base + pos] = t;
            tok_w [(size_t)e * T_TOK + base + pos] = w;
        }
        __syncthreads();
        if (tid == 0) base_s = base + total;
        __syncthreads();
    }
    if (tid == 0) cnt[e] = base_s;
}

// ---------------- aux loss + counts (deterministic reduction) ----------------
__global__ __launch_bounds__(256) void finalize_kernel(
    const float* __restrict__ probs, const int* __restrict__ cnt, float* __restrict__ out_tail)
{
    __shared__ float red[256];
    __shared__ float Ps[NEXP];
    const int tid = threadIdx.x;
    float acc[NEXP];
#pragma unroll
    for (int e = 0; e < NEXP; ++e) acc[e] = 0.0f;
    for (int t = tid; t < T_TOK; t += 256) {
#pragma unroll
        for (int e = 0; e < NEXP; ++e) acc[e] += probs[(size_t)t * NEXP + e];
    }
    for (int e = 0; e < NEXP; ++e) {
        red[tid] = acc[e];
        __syncthreads();
        for (int off = 128; off > 0; off >>= 1) {
            if (tid < off) red[tid] += red[tid + off];
            __syncthreads();
        }
        if (tid == 0) Ps[e] = red[0] / (float)T_TOK;
        __syncthreads();
    }
    if (tid == 0) {
        float aux = 0.0f;
        for (int e = 0; e < NEXP; ++e) {
            float f = (float)cnt[e] / (float)(2 * T_TOK);
            aux += f * Ps[e];
        }
        out_tail[0] = (float)NEXP * aux;
    }
    if (tid < NEXP) out_tail[1 + tid] = (float)cnt[tid];
}

// ---------------- expert FFN: WMMA bf16, 16-token M-tile per block ----------------
__global__ __launch_bounds__(256) void expert_ffn_kernel(
    const __bf16* __restrict__ xbf, const __bf16* __restrict__ wgate,
    const __bf16* __restrict__ wup, const __bf16* __restrict__ wdown,
    const int* __restrict__ tok_idx, const float* __restrict__ tok_w,
    const int* __restrict__ cnt, float* __restrict__ out)
{
    const int e  = blockIdx.y;
    const int m0 = blockIdx.x * 16;
    const int count = cnt[e];
    if (m0 >= count) return;                 // uniform per block

    __shared__ __bf16 Act[16][ACT_STRIDE];   // SwiGLU activations (phase-2 A matrix)
    __shared__ int   row_tok[16];
    __shared__ int   row_safe[16];
    __shared__ float row_w[16];

    const int tid  = threadIdx.x;
    const int wv   = tid >> 5;
    const int lane = tid & 31;

    if (tid < 16) {
        int gm = m0 + tid;
        int t = -1; float w = 0.0f;
        if (gm < count) {
            t = tok_idx[(size_t)e * T_TOK + gm];
            w = tok_w [(size_t)e * T_TOK + gm];
        }
        row_tok[tid]  = t;
        row_safe[tid] = (t < 0) ? 0 : t;     // garbage rows computed but never stored
        row_w[tid]    = w;
    }
    __syncthreads();

    const int n     = lane & 15;             // A-row / B-col / C-col for this lane
    const int koffA = (lane >> 4) << 4;      // A: 0 or 16 (K 0..15 / 16..31 halves)
    const int koffB = (lane >> 4) << 4;      // B: 0 or 16 (K 0..15 / 16..31 halves)
    const int mbase = (lane >> 4) << 3;      // C/D: rows v or v+8

    const size_t wbofs = (size_t)e * (size_t)DFF * (size_t)DMODEL;
    const __bf16* arow = xbf + (size_t)row_safe[n] * DMODEL;

    // ---------- phase 1: g = x*Wg^T, u = x*Wu^T, act = silu(g)*u ----------
    for (int i = 0; i < 8; ++i) {
        const int nt = wv + (i << 3);        // 64 N-tiles over DFF split across 8 waves
        const __bf16* bg = wgate + wbofs + (size_t)(nt * 16 + n) * DMODEL;
        const __bf16* bu = wup   + wbofs + (size_t)(nt * 16 + n) * DMODEL;
        v8f accg = {}; v8f accu = {};
        for (int k0 = 0; k0 < DMODEL; k0 += 32) {
            Frag a, fg, fu;
            a.u[0]  = *(const u32x4*)(arow + k0 + koffA);
            a.u[1]  = *(const u32x4*)(arow + k0 + koffA + 8);
            fg.u[0] = *(const u32x4*)(bg + k0 + koffB);
            fg.u[1] = *(const u32x4*)(bg + k0 + koffB + 8);
            fu.u[0] = *(const u32x4*)(bu + k0 + koffB);
            fu.u[1] = *(const u32x4*)(bu + k0 + koffB + 8);
            accg = __builtin_amdgcn_wmma_f32_16x16x32_bf16(false, a.v, false, fg.v,
                                                           (short)0, accg, false, false);
            accu = __builtin_amdgcn_wmma_f32_16x16x32_bf16(false, a.v, false, fu.v,
                                                           (short)0, accu, false, false);
        }
#pragma unroll
        for (int v = 0; v < 8; ++v) {
            float g = accg[v], u = accu[v];
            float act = (g / (1.0f + __expf(-g))) * u;   // silu(g)*u
            Act[v + mbase][nt * 16 + n] = (__bf16)act;
        }
    }
    __syncthreads();

    // ---------- phase 2: out_tile = act * Wd^T, weighted scatter ----------
    for (int i = 0; i < 8; ++i) {
        const int nt = wv + (i << 3);        // 64 N-tiles over DMODEL
        const __bf16* bd = wdown + wbofs + (size_t)(nt * 16 + n) * DFF;
        v8f acc = {};
        for (int k0 = 0; k0 < DFF; k0 += 32) {
            Frag a, fb;
            a.u[0]  = *(const u32x4*)&Act[n][k0 + koffA];
            a.u[1]  = *(const u32x4*)&Act[n][k0 + koffA + 8];
            fb.u[0] = *(const u32x4*)(bd + k0 + koffB);
            fb.u[1] = *(const u32x4*)(bd + k0 + koffB + 8);
            acc = __builtin_amdgcn_wmma_f32_16x16x32_bf16(false, a.v, false, fb.v,
                                                          (short)0, acc, false, false);
        }
#pragma unroll
        for (int v = 0; v < 8; ++v) {
            int m = v + mbase;
            if (m0 + m < count) {
                int t = row_tok[m];
                float val = acc[v] * row_w[m];
                __hip_atomic_fetch_add(&out[(size_t)t * DMODEL + nt * 16 + n], val,
                                       __ATOMIC_RELAXED, __HIP_MEMORY_SCOPE_AGENT);
            }
        }
    }
}

// ---------------- launch ----------------
extern "C" void kernel_launch(void* const* d_in, const int* in_sizes, int n_in,
                              void* d_out, int out_size, void* d_ws, size_t ws_size,
                              hipStream_t stream) {
    (void)in_sizes; (void)n_in; (void)out_size; (void)ws_size;
    const float* x      = (const float*)d_in[0];
    const float* gate_w = (const float*)d_in[1];
    const float* w_gate = (const float*)d_in[2];
    const float* w_up   = (const float*)d_in[3];
    const float* w_down = (const float*)d_in[4];
    float* out = (float*)d_out;

    const int NTOT = T_TOK * DMODEL;              // 8,388,608
    const int NW   = NEXP * DFF * DMODEL;         // 8,388,608

    char* ws = (char*)d_ws;
    size_t o = 0;
    __bf16* xbf   = (__bf16*)(ws + o); o += (size_t)NTOT * 2;
    __bf16* wg_bf = (__bf16*)(ws + o); o += (size_t)NW * 2;
    __bf16* wu_bf = (__bf16*)(ws + o); o += (size_t)NW * 2;
    __bf16* wd_bf = (__bf16*)(ws + o); o += (size_t)NW * 2;
    float* probs  = (float*)(ws + o);  o += (size_t)T_TOK * NEXP * 4;
    int*   comb_e = (int*)(ws + o);    o += (size_t)T_TOK * 2 * 4;
    float* comb_w = (float*)(ws + o);  o += (size_t)T_TOK * 2 * 4;
    int*   tok_idx= (int*)(ws + o);    o += (size_t)NEXP * T_TOK * 4;
    float* tok_w  = (float*)(ws + o);  o += (size_t)NEXP * T_TOK * 4;
    int*   cnt    = (int*)(ws + o);    o += 64;

    zero_kernel<<<4096, 256, 0, stream>>>(out, NTOT);
    cvt_bf16_kernel<<<4096, 256, 0, stream>>>(xbf,   x,      NTOT);
    cvt_bf16_kernel<<<4096, 256, 0, stream>>>(wg_bf, w_gate, NW);
    cvt_bf16_kernel<<<4096, 256, 0, stream>>>(wu_bf, w_up,   NW);
    cvt_bf16_kernel<<<4096, 256, 0, stream>>>(wd_bf, w_down, NW);

    router_kernel<<<T_TOK / 8, 256, 0, stream>>>(x, gate_w, probs, comb_e, comb_w);
    compact_kernel<<<NEXP, 256, 0, stream>>>(comb_e, comb_w, tok_idx, tok_w, cnt);
    finalize_kernel<<<1, 256, 0, stream>>>(probs, cnt, out + (size_t)NTOT);

    expert_ffn_kernel<<<dim3(T_TOK / 16, NEXP), 256, 0, stream>>>(
        xbf, wg_bf, wu_bf, wd_bf, tok_idx, tok_w, cnt, out);
}
